// CGCNNModel_74156905332881
// MI455X (gfx1250) — compile-verified, matrix operand order checked by reading, not run on hardware
//
#include <hip/hip_runtime.h>
#include <hip/hip_bf16.h>

// ---------------- problem constants ----------------
#define B_     16
#define N_     1024
#define M_     12
#define F_     64
#define BF_    128
#define INFEA  256              // 2F + BF
#define NCOLT  5                // 5 col-tiles of 16: 64 core + 1 filt + 15 pad
#define TILE_A 8                // atoms per workgroup
#define ROWS   96               // TILE_A * M_  (== 6 tiles of 16, exact)
#define MT     6                // row tiles per workgroup
#define RSTR   264              // ushort stride per staged row (256 + 8 pad)
#define CSTR   65               // float stride for core scratch
#define BN_INV 0.99950038f      // 1/sqrt(1 + 1e-3)
#define PACK_L (8 * NCOLT * 32 * 16)   // bf16 elems per layer of packed W

typedef __attribute__((ext_vector_type(16))) __bf16 v16bf;
typedef __attribute__((ext_vector_type(8)))  float  v8f;

union Frag { uint4 q[2]; v16bf v; };

__device__ __forceinline__ unsigned short f2bf(float f) {
    unsigned u = __float_as_uint(f);
    u += 0x7FFFu + ((u >> 16) & 1u);          // round-to-nearest-even
    return (unsigned short)(u >> 16);
}

// async global->LDS copy, 16B per lane; LDS dest = byte offset in group segment
__device__ __forceinline__ void async_g2l_b128(unsigned lds_off, const void* gptr) {
    asm volatile("global_load_async_to_lds_b128 %0, %1, off"
                 :: "v"(lds_off), "v"(gptr) : "memory");
}
__device__ __forceinline__ void wait_asynccnt0() {
#if __has_builtin(__builtin_amdgcn_s_wait_asynccnt)
    __builtin_amdgcn_s_wait_asynccnt(0);
#else
    asm volatile("s_wait_asynccnt 0x0" ::: "memory");
#endif
}

// ---------------- weight repack: [l][kt][nt][lane][e] bf16 B-fragments ----
__global__ void pack_w_kernel(const float* __restrict__ core_w,
                              const float* __restrict__ filt_w,
                              unsigned short* __restrict__ wp) {
    int idx = blockIdx.x * 256 + threadIdx.x;
    if (idx >= 3 * PACK_L) return;
    int e    = idx & 15;
    int lane = (idx >> 4) & 31;
    int nt   = (idx >> 9) % NCOLT;
    int q    = (idx >> 9) / NCOLT;           // l*8 + kt
    int kt   = q & 7;
    int l    = q >> 3;
    int col  = nt * 16 + (lane & 15);
    int K    = kt * 32 + ((lane & 16) ? 8 : 0) + (e < 8 ? e : e + 8);
    float v = 0.f;
    if (col < 64)       v = core_w[(l * INFEA + K) * F_ + col];
    else if (col == 64) v = filt_w[l * INFEA + K];
    wp[idx] = f2bf(v);
}

// ---------------- embedding gather ----------------
__global__ void embed_kernel(const int* __restrict__ atom_types,
                             const float* __restrict__ emb,
                             float* __restrict__ x0) {
    int idx = blockIdx.x * 256 + threadIdx.x;      // B*N*F elems
    int atom = idx >> 6, f = idx & 63;
    x0[idx] = emb[atom_types[atom] * F_ + f];
}

// ---------------- one CGCNN conv layer ----------------
// bb_in  != null: stage bond rows from bf16 buffer via async-to-LDS copies
// bb_in  == null: stage from fp32 bond with on-the-fly convert
// bb_out != null: additionally persist converted bf16 bond rows (layer 0)
__global__ __launch_bounds__(256) void cgcnn_layer_kernel(
    const float* __restrict__ xin, float* __restrict__ xout,
    const float* __restrict__ bond, const int* __restrict__ nbr,
    const unsigned short* __restrict__ bb_in, unsigned short* __restrict__ bb_out,
    const unsigned short* __restrict__ wp,
    const float* __restrict__ cb,  const float* __restrict__ bag,
    const float* __restrict__ bab, const float* __restrict__ bbg,
    const float* __restrict__ bbb) {
    __shared__ __align__(16) unsigned char smem[ROWS * RSTR * 2];   // 50688 B
    unsigned short* totalS = (unsigned short*)smem;
    float* coreS = (float*)smem;                       // 96*65*4 = 24960 B
    float* filtS = (float*)(smem + ROWS * CSTR * 4);   // +384 B (aliases stage buf)

    const int tid  = threadIdx.x;
    const int lane = tid & 31;
    const int wave = tid >> 5;
    const int b    = blockIdx.y;
    const int n0   = blockIdx.x * TILE_A;

    // ---- phase 1: stage total = [self | nbr | bond] rows as bf16 in LDS ----
    if (bb_in) {
        // bond half: async bf16 copies straight into LDS (16B per lane)
        for (int c = tid; c < ROWS * 16; c += 256) {   // 16x 16B chunks per row
            int row = c >> 4, j = c & 15;
            int a = row / 12, m = row - a * 12;
            unsigned lds_off = (unsigned)(row * (RSTR * 2) + 2 * F_ * 2 + j * 16);
            const unsigned short* g =
                bb_in + (size_t)((b * N_ + (n0 + a)) * M_ + m) * BF_ + j * 8;
            async_g2l_b128(lds_off, g);
        }
    }
    for (int i = tid; i < ROWS * F_; i += 256) {
        int row = i >> 6, f = i & 63;
        int a = row / 12, m = row - a * 12;
        int n = n0 + a;
        totalS[row * RSTR + f] = f2bf(xin[(b * N_ + n) * F_ + f]);
        int nb = nbr[(b * N_ + n) * M_ + m];
        totalS[row * RSTR + F_ + f] = f2bf(xin[(b * N_ + nb) * F_ + f]);
    }
    if (!bb_in) {
        for (int i = tid; i < ROWS * BF_; i += 256) {
            int row = i >> 7, j = i & 127;
            int a = row / 12, m = row - a * 12;
            size_t gidx = (size_t)((b * N_ + (n0 + a)) * M_ + m) * BF_ + j;
            unsigned short v = f2bf(bond[gidx]);
            totalS[row * RSTR + 2 * F_ + j] = v;
            if (bb_out) bb_out[gidx] = v;              // persist bf16 for layers 1,2
        }
    } else {
        wait_asynccnt0();
    }
    __syncthreads();

    // ---- phase 2: WMMA GEMM, wave w owns M-tile w ----
    v8f acc[NCOLT];
    const int mt = wave;
    if (wave < MT) {
        Frag af[8];
        const int rowA = mt * 16 + (lane & 15);
        const int hi   = (lane >> 4) & 1;               // K +8 half for hi lanes
        const unsigned short* rp = totalS + rowA * RSTR + hi * 8;
#pragma unroll
        for (int kt = 0; kt < 8; ++kt) {                // two ds_load_b128 each
            const uint4* p = (const uint4*)(rp + kt * 32);
            af[kt].q[0] = p[0];
            af[kt].q[1] = p[2];                         // +16 ushorts
        }
#pragma unroll
        for (int nt = 0; nt < NCOLT; ++nt) {
            v8f c = {};
#pragma unroll
            for (int kt = 0; kt < 8; ++kt) {
                Frag bf_;
                const uint4* gp =
                    (const uint4*)(wp + ((kt * NCOLT + nt) * 32 + lane) * 16);
                bf_.q[0] = gp[0];
                bf_.q[1] = gp[1];
                c = __builtin_amdgcn_wmma_f32_16x16x32_bf16(
                        false, af[kt].v, false, bf_.v, (short)0, c, false, false);
            }
            acc[nt] = c;
        }
    }
    __syncthreads();   // all LDS reads done -> safe to overwrite staging buffer

    if (wave < MT) {
        const int hi8 = ((lane >> 4) & 1) << 3;
#pragma unroll
        for (int nt = 0; nt < 4; ++nt) {
            int colg = nt * 16 + (lane & 15);
            float bias = cb[colg];
#pragma unroll
            for (int v = 0; v < 8; ++v) {
                int row = mt * 16 + v + hi8;
                coreS[row * CSTR + colg] = acc[nt][v] + bias;
            }
        }
        if ((lane & 15) == 0) {                         // filt column (bias drops in softmax)
#pragma unroll
            for (int v = 0; v < 8; ++v) {
                int row = mt * 16 + v + hi8;
                filtS[row] = acc[4][v];
            }
        }
    }
    __syncthreads();

    // ---- phase 3: per-atom softmax/BN/relu/residual, one wave per atom ----
    {
        const int a = wave;
        const int n = n0 + a;
        float mx = -1e30f;
#pragma unroll
        for (int m = 0; m < M_; ++m) mx = fmaxf(mx, filtS[a * M_ + m]);
        float s = 0.f;
#pragma unroll
        for (int m = 0; m < M_; ++m) s += __expf(filtS[a * M_ + m] - mx);
        float inv = 1.f / s;
#pragma unroll
        for (int h = 0; h < 2; ++h) {
            int f = lane + h * 32;
            float g1 = bag[f], b1 = bab[f];
            float sum = 0.f;
#pragma unroll
            for (int m = 0; m < M_; ++m) {
                int row = a * M_ + m;
                float w = __expf(filtS[row] - mx) * inv;
                float c = coreS[row * CSTR + f];
                c = g1 * (c * BN_INV) + b1;             // BN-a
                c = fmaxf(c, 0.f);                      // relu
                sum += w * c;
            }
            sum *= (1.f / 12.f);                        // mean over neighbors
            float s2 = bbg[f] * (sum * BN_INV) + bbb[f]; // BN-b
            xout[(b * N_ + n) * F_ + f] =
                fmaxf(xin[(b * N_ + n) * F_ + f] + s2, 0.f);
        }
    }
}

// ---------------- pooling gather + dense head ----------------
__global__ void pool_dense_kernel(const float* __restrict__ x,
                                  const int* __restrict__ tgt,
                                  const float* __restrict__ dw,
                                  const float* __restrict__ db,
                                  float* __restrict__ out) {
    __shared__ float xv[F_];
    int b = blockIdx.y, t = blockIdx.x, f = threadIdx.x;
    int ti = tgt[b * 64 + t];
    xv[f] = fmaxf(x[(b * N_ + ti) * F_ + f], 0.f);
    __syncthreads();
    float acc = db[f];
#pragma unroll 8
    for (int k = 0; k < F_; ++k) acc += xv[k] * dw[k * F_ + f];
    out[(b * 64 + t) * F_ + f] = fmaxf(acc, 0.f);
}

// ---------------- launcher ----------------
extern "C" void kernel_launch(void* const* d_in, const int* in_sizes, int n_in,
                              void* d_out, int out_size, void* d_ws, size_t ws_size,
                              hipStream_t stream) {
    const int*   atom_types = (const int*)  d_in[0];
    const float* bond       = (const float*)d_in[1];
    const int*   nbr        = (const int*)  d_in[2];
    const int*   tgt        = (const int*)  d_in[3];
    const float* emb        = (const float*)d_in[4];
    const float* core_w     = (const float*)d_in[5];
    const float* core_b     = (const float*)d_in[6];
    const float* filt_w     = (const float*)d_in[7];
    /* filt_b (d_in[8]) cancels in softmax */
    const float* bna_g      = (const float*)d_in[9];
    const float* bna_b      = (const float*)d_in[10];
    const float* bnb_g      = (const float*)d_in[11];
    const float* bnb_b      = (const float*)d_in[12];
    const float* dense_w    = (const float*)d_in[13];
    const float* dense_b    = (const float*)d_in[14];
    float* out = (float*)d_out;

    const size_t XSZ    = (size_t)B_ * N_ * F_;            // 1,048,576 floats
    const size_t BONDSZ = (size_t)B_ * N_ * M_ * BF_;      // 25,165,824 elems
    float* xA = (float*)d_ws;
    float* xB = xA + XSZ;
    unsigned short* wpack  = (unsigned short*)(xB + XSZ);
    unsigned short* bondbf = wpack + (size_t)3 * PACK_L;
    const size_t need = 2 * XSZ * 4 + (size_t)3 * PACK_L * 2 + BONDSZ * 2;
    const bool big = (ws_size >= need);                    // bf16-bond cache fits?

    pack_w_kernel<<<(3 * PACK_L + 255) / 256, 256, 0, stream>>>(core_w, filt_w, wpack);
    embed_kernel<<<(int)(XSZ / 256), 256, 0, stream>>>(atom_types, emb, xA);

    dim3 lgrid(N_ / TILE_A, B_);
    const float* xi[3] = { xA, xB, xA };
    float*       xo[3] = { xB, xA, xB };
    for (int l = 0; l < 3; ++l) {
        const unsigned short* bbin  = (big && l > 0)  ? bondbf : nullptr;
        unsigned short*       bbout = (big && l == 0) ? bondbf : nullptr;
        cgcnn_layer_kernel<<<lgrid, 256, 0, stream>>>(
            xi[l], xo[l], bond, nbr, bbin, bbout,
            wpack + (size_t)l * PACK_L,
            core_b + l * F_, bna_g + l * F_, bna_b + l * F_,
            bnb_g + l * F_, bnb_b + l * F_);
    }

    dim3 dgrid(64, B_);
    pool_dense_kernel<<<dgrid, F_, 0, stream>>>(xB, tgt, dense_w, dense_b, out);
}
